// LstmCRFModel_22900765623081
// MI455X (gfx1250) — compile-verified, matrix operand order checked by reading, not run on hardware
//
#include <hip/hip_runtime.h>

// ---------------- problem constants ----------------
#define B_   512
#define T_   128
#define D_   128
#define H_   256
#define G4   1024      // 4*H
#define KH   384       // D + H  (combined GEMM K for the recurrence)
#define KL   49        // CRF labels
#define LL   7         // CRF sequence length
#define E2H  512       // 2*H (bi-LSTM feature width)
#define NFC  343       // K*L fc output columns
#define NTFC 22        // ceil(343/16) N tiles for fc

typedef __bf16 bf16_t;
typedef __attribute__((ext_vector_type(16))) __bf16 v16bf;
typedef __attribute__((ext_vector_type(8)))  __bf16 v8bf;
typedef __attribute__((ext_vector_type(8)))  float  v8f;
// TDM descriptor operand types (clang-23 6-arg builtin form)
typedef __attribute__((ext_vector_type(4))) unsigned int tdm_v4u;
typedef __attribute__((ext_vector_type(8))) int          tdm_v8i;
typedef __attribute__((ext_vector_type(4))) int          tdm_v4i;

union BFU { unsigned short u; bf16_t b; };

__device__ __forceinline__ bf16_t f2bf(float f) {
  unsigned u = __float_as_uint(f);
  unsigned r = (u + 0x7FFFu + ((u >> 16) & 1u)) >> 16;   // RNE
  BFU x; x.u = (unsigned short)r; return x.b;
}
__device__ __forceinline__ float bf2f(bf16_t b) {
  BFU x; x.b = b; return __uint_as_float(((unsigned)x.u) << 16);
}

union FRAG { v16bf v; v8bf h[2]; };

// ISA 16-bit A/B fragment layout (16x32 / 32x16):
// lane half (lane>>4) selects the K octet; elements 0..7 = K[k0+half*8 .. +7],
// elements 8..15 = K[k0+16+half*8 .. +7]. Row (M for A, N for B) = lane&15,
// with the matrix stored row-major [row][K] (B is W^T, W row-major [N][K]).
__device__ __forceinline__ v16bf load_frag(const bf16_t* base, int ld, int row, int k0) {
  const int lane = threadIdx.x & 31;
  const int half = lane >> 4;
  const bf16_t* p = base + (size_t)row * ld + k0 + half * 8;
  FRAG f;
  f.h[0] = *(const v8bf*)p;
  f.h[1] = *(const v8bf*)(p + 16);
  return f.v;
}
__device__ __forceinline__ v16bf zero_frag() {
  FRAG f;
  #pragma unroll
  for (int i = 0; i < 16; ++i) f.v[i] = f2bf(0.0f);
  return f.v;
}

__device__ __forceinline__ float sigf(float x) { return 1.0f / (1.0f + __expf(-x)); }

// ---------------- conversion / packing ----------------
__global__ void k_init(float* out) { if (threadIdx.x == 0) out[0] = 0.0f; }

__global__ void k_cvt(const float* __restrict__ s, bf16_t* __restrict__ d, int n) {
  int i = blockIdx.x * 256 + threadIdx.x;
  if (i < n) d[i] = f2bf(s[i]);
}

// Wc[dir][g][k] : k<128 -> w_ih[g][k], else w_hh[g][k-128]
__global__ void k_pack_wc(const float* __restrict__ wih_f, const float* __restrict__ whh_f,
                          const float* __restrict__ wih_b, const float* __restrict__ whh_b,
                          bf16_t* __restrict__ wc) {
  int i = blockIdx.x * 256 + threadIdx.x;
  const int per = G4 * KH;
  if (i >= 2 * per) return;
  int dir = i / per, rem = i % per;
  int g = rem / KH, k = rem % KH;
  const float* wih = dir ? wih_b : wih_f;
  const float* whh = dir ? whh_b : whh_f;
  float v = (k < D_) ? wih[g * D_ + k] : whh[g * H_ + (k - D_)];
  wc[i] = f2bf(v);
}

// ---------------- BiLSTM recurrence ----------------
// grid (B/16, 2), block 256 (8 waves). Each WG owns 16 batch rows for one
// direction; c stays in f32 registers; h round-trips through LDS as bf16.
// x_t tile (16x128 bf16, row stride T*D) is staged into LDS by the TDM with
// per-row padding so rows land at the 384-element a_stage pitch.
__global__ __launch_bounds__(256)
void k_lstm(const bf16_t* __restrict__ xbf, const bf16_t* __restrict__ wc,
            const float* __restrict__ b_f, const float* __restrict__ b_b,
            bf16_t* __restrict__ lstm_out) {
  const int dir = blockIdx.y;
  const int b0  = blockIdx.x * 16;
  const bf16_t* Wc  = wc + (size_t)dir * G4 * KH;        // [1024][384]
  const float* bias = dir ? b_b : b_f;

  __shared__ bf16_t a_stage[16][KH];   // cols 0..127: x_t ; 128..383: h_{t-1}

  const int tid  = threadIdx.x;
  const int lane = tid & 31;
  const int wid  = tid >> 5;
  const int half = lane >> 4;
  const int ln   = lane & 15;
  const int jt0  = wid, jt1 = wid + 8; // two hidden 16-col tiles per wave

  for (int i = tid; i < 16 * H_; i += 256)
    a_stage[i >> 8][128 + (i & 255)] = f2bf(0.0f);       // h0 = 0

  float c0[8], c1[8];
  #pragma unroll
  for (int r = 0; r < 8; ++r) { c0[r] = 0.0f; c1[r] = 0.0f; }

  const unsigned lds_base = (unsigned)(size_t)&a_stage[0][0];
  const unsigned long long gxbase =
      (unsigned long long)(size_t)xbf + (unsigned long long)b0 * T_ * D_ * 2ull;

  for (int step = 0; step < T_; ++step) {
    const int tsrc = dir ? (T_ - 1 - step) : step;

    // stage x_t via Tensor Data Mover (one wave issues; TENSORcnt is per-wave,
    // so the issuing wave drains it before signaling the workgroup barrier)
    if (wid == 0) {
      const unsigned long long ga = gxbase + (unsigned long long)tsrc * (D_ * 2);
      tdm_v4u g0;
      g0[0] = 1u;                                   // count=1, user mode
      g0[1] = lds_base;                             // lds_addr
      g0[2] = (unsigned)ga;                         // global_addr[31:0]
      g0[3] = (unsigned)(ga >> 32) | 0x80000000u;   // global_addr[56:32] | type=2
      tdm_v8i g1;
      g1[0] = (int)0xFF510000u;   // data_size=2B, pad_enable, pad_int=64dw, pad_amt=128dw
      g1[1] = (int)(128u << 16);  // tensor_dim0 = 128
      g1[2] = (int)(16u << 16);   // tensor_dim1 = 16
      g1[3] = (int)(128u << 16);  // tile_dim0 = 128
      g1[4] = 16;                 // tile_dim1 = 16
      g1[5] = T_ * D_;            // tensor_dim0_stride = 16384 elements
      g1[6] = 0; g1[7] = 0;
      tdm_v4i zz4; zz4[0] = 0; zz4[1] = 0; zz4[2] = 0; zz4[3] = 0;
      tdm_v8i zz8;
      #pragma unroll
      for (int i = 0; i < 8; ++i) zz8[i] = 0;
      __builtin_amdgcn_tensor_load_to_lds(g0, g1, zz4, zz4, zz8, 0);
      __builtin_amdgcn_s_wait_tensorcnt(0);
    }
    __syncthreads();

    v8f acc[2][4];
    #pragma unroll
    for (int q = 0; q < 2; ++q)
      #pragma unroll
      for (int g = 0; g < 4; ++g)
        #pragma unroll
        for (int r = 0; r < 8; ++r) acc[q][g][r] = 0.0f;

    #pragma unroll 1
    for (int kt = 0; kt < KH / 32; ++kt) {
      const int k0 = kt * 32;
      v16bf a = load_frag(&a_stage[0][0], KH, ln, k0);
      // issue all 8 weight fragments before any WMMA: 16 b128 loads in flight
      v16bf bfr[8];
      #pragma unroll
      for (int q = 0; q < 2; ++q)
        #pragma unroll
        for (int g = 0; g < 4; ++g) {
          const int jt = q ? jt1 : jt0;
          bfr[q * 4 + g] = load_frag(Wc, KH, g * H_ + jt * 16 + ln, k0);
        }
      if (kt + 1 < KH / 32)
        __builtin_prefetch(Wc + (size_t)(jt0 * 16 + ln) * KH + k0 + 32, 0, 1);
      #pragma unroll
      for (int q = 0; q < 2; ++q)
        #pragma unroll
        for (int g = 0; g < 4; ++g)
          acc[q][g] = __builtin_amdgcn_wmma_f32_16x16x32_bf16(
              false, a, false, bfr[q * 4 + g], (short)0, acc[q][g], false, false);
    }
    __syncthreads();   // all waves done reading x and h_{t-1}

    #pragma unroll
    for (int q = 0; q < 2; ++q) {
      const int jt = q ? jt1 : jt0;
      const int j  = jt * 16 + ln;                  // hidden column 0..255
      const float bi = bias[0 * H_ + j];
      const float bfv = bias[1 * H_ + j];
      const float bg = bias[2 * H_ + j];
      const float bo = bias[3 * H_ + j];
      float* cc = q ? c1 : c0;
      #pragma unroll
      for (int r = 0; r < 8; ++r) {
        const int m = r + 8 * half;
        float gi = sigf(acc[q][0][r] + bi);
        float gf = sigf(acc[q][1][r] + bfv);
        float gg = tanhf(acc[q][2][r] + bg);
        float go = sigf(acc[q][3][r] + bo);
        float cn = gf * cc[r] + gi * gg;
        cc[r] = cn;
        bf16_t hb = f2bf(go * tanhf(cn));
        a_stage[m][128 + j] = hb;                   // feed next step
        lstm_out[((size_t)(b0 + m) * T_ + tsrc) * E2H + dir * H_ + j] = hb;
      }
    }
    __syncthreads();   // h_t visible before next step's GEMM
  }
}

// ---------------- attention: a = tanh(lstm_out @ aw1^T + ab1) ----------------
__global__ __launch_bounds__(256)
void k_att_a(const bf16_t* __restrict__ lstm_out, const bf16_t* __restrict__ aw1b,
             const float* __restrict__ ab1, bf16_t* __restrict__ a_out) {
  const int wgid = blockIdx.x * 8 + (threadIdx.x >> 5);
  const int mt = wgid >> 4;          // 0..4095
  const int nt = wgid & 15;          // 0..15
  const int lane = threadIdx.x & 31;
  const int half = lane >> 4, ln = lane & 15;

  v8f acc;
  #pragma unroll
  for (int r = 0; r < 8; ++r) acc[r] = 0.0f;

  #pragma unroll 1
  for (int kt = 0; kt < E2H / 32; ++kt) {
    v16bf a = load_frag(lstm_out, E2H, mt * 16 + ln, kt * 32);
    v16bf b = load_frag(aw1b,     E2H, nt * 16 + ln, kt * 32);
    acc = __builtin_amdgcn_wmma_f32_16x16x32_bf16(
        false, a, false, b, (short)0, acc, false, false);
  }
  const int n = nt * 16 + ln;
  const float bb = ab1[n];
  #pragma unroll
  for (int r = 0; r < 8; ++r) {
    const int m = mt * 16 + r + 8 * half;
    a_out[(size_t)m * H_ + n] = f2bf(tanhf(acc[r] + bb));
  }
}

// scores[row] = dot(a[row], aw2) + ab2
__global__ __launch_bounds__(256)
void k_score(const bf16_t* __restrict__ a_out, const float* __restrict__ aw2,
             const float* __restrict__ ab2, float* __restrict__ scores) {
  const int row  = blockIdx.x * 8 + (threadIdx.x >> 5);
  const int lane = threadIdx.x & 31;
  float s = 0.0f;
  #pragma unroll
  for (int i = lane; i < H_; i += 32) s += bf2f(a_out[(size_t)row * H_ + i]) * aw2[i];
  #pragma unroll
  for (int off = 16; off; off >>= 1) s += __shfl_xor(s, off, 32);
  if (lane == 0) scores[row] = s + ab2[0];
}

// per-b softmax over T + weighted feature pooling
__global__ __launch_bounds__(128)
void k_pool(const float* __restrict__ scores, const bf16_t* __restrict__ lstm_out,
            bf16_t* __restrict__ feat) {
  __shared__ float red[128];
  __shared__ float w[128];
  const int b = blockIdx.x, tid = threadIdx.x;
  float s = scores[b * T_ + tid];
  red[tid] = s; __syncthreads();
  for (int off = 64; off; off >>= 1) {
    if (tid < off) red[tid] = fmaxf(red[tid], red[tid + off]);
    __syncthreads();
  }
  float mx = red[0]; __syncthreads();
  float e = __expf(s - mx);
  red[tid] = e; __syncthreads();
  for (int off = 64; off; off >>= 1) {
    if (tid < off) red[tid] += red[tid + off];
    __syncthreads();
  }
  float inv = 1.0f / red[0];
  w[tid] = e * inv; __syncthreads();
  for (int e0 = tid; e0 < E2H; e0 += 128) {
    float acc = 0.0f;
    for (int t = 0; t < T_; ++t)
      acc += w[t] * bf2f(lstm_out[((size_t)b * T_ + t) * E2H + e0]);
    feat[(size_t)b * E2H + e0] = f2bf(acc);
  }
}

// em[512][343] = feat @ fc_w^T + fc_b
__global__ __launch_bounds__(256)
void k_fc(const bf16_t* __restrict__ feat, const bf16_t* __restrict__ fcwb,
          const float* __restrict__ fc_b, float* __restrict__ em) {
  const int wgid = blockIdx.x * 8 + (threadIdx.x >> 5);   // 0..703 exactly
  const int mt = wgid / NTFC, nt = wgid % NTFC;
  const int lane = threadIdx.x & 31;
  const int half = lane >> 4, ln = lane & 15;
  const int n = nt * 16 + ln;

  v8f acc;
  #pragma unroll
  for (int r = 0; r < 8; ++r) acc[r] = 0.0f;

  #pragma unroll 1
  for (int kt = 0; kt < E2H / 32; ++kt) {
    v16bf a = load_frag(feat, E2H, mt * 16 + ln, kt * 32);
    v16bf b = (n < NFC) ? load_frag(fcwb, E2H, n, kt * 32) : zero_frag();
    acc = __builtin_amdgcn_wmma_f32_16x16x32_bf16(
        false, a, false, b, (short)0, acc, false, false);
  }
  if (n < NFC) {
    const float bb = fc_b[n];
    #pragma unroll
    for (int r = 0; r < 8; ++r) {
      const int m = mt * 16 + r + 8 * half;
      em[(size_t)m * NFC + n] = acc[r] + bb;
    }
  }
}

// ---------------- CRF forward + gold score ----------------
__global__ __launch_bounds__(64)
void k_crf(const float* __restrict__ em, const int* __restrict__ labels,
           const float* __restrict__ start_t, const float* __restrict__ trans,
           const float* __restrict__ end_t, float* __restrict__ out) {
  __shared__ float alpha[KL];
  __shared__ float alpha2[KL];
  const int b = blockIdx.x, j = threadIdx.x;
  const float* emb = em + (size_t)b * (LL * KL);

  if (j < KL) alpha[j] = start_t[j] + emb[j];
  __syncthreads();
  for (int l = 1; l < LL; ++l) {
    if (j < KL) {
      float mx = -3.4e38f;
      for (int i = 0; i < KL; ++i) mx = fmaxf(mx, alpha[i] + trans[i * KL + j]);
      float s = 0.0f;
      for (int i = 0; i < KL; ++i) s += __expf(alpha[i] + trans[i * KL + j] - mx);
      alpha2[j] = mx + __logf(s) + emb[l * KL + j];
    }
    __syncthreads();
    if (j < KL) alpha[j] = alpha2[j];
    __syncthreads();
  }
  if (j == 0) {
    float mx = -3.4e38f;
    for (int i = 0; i < KL; ++i) mx = fmaxf(mx, alpha[i] + end_t[i]);
    float s = 0.0f;
    for (int i = 0; i < KL; ++i) s += __expf(alpha[i] + end_t[i] - mx);
    float logZ = mx + __logf(s);

    const int* lab = labels + b * LL;
    float sc = start_t[lab[0]] + end_t[lab[LL - 1]];
    for (int l = 0; l < LL; ++l) sc += emb[l * KL + lab[l]];
    for (int l = 0; l < LL - 1; ++l) sc += trans[lab[l] * KL + lab[l + 1]];

    atomicAdd(out, (logZ - sc) * (1.0f / (float)B_));
  }
}

// ---------------- host launcher ----------------
extern "C" void kernel_launch(void* const* d_in, const int* in_sizes, int n_in,
                              void* d_out, int out_size, void* d_ws, size_t ws_size,
                              hipStream_t stream) {
  (void)in_sizes; (void)n_in; (void)out_size; (void)ws_size;
  const float* x       = (const float*)d_in[0];
  const int*   labels  = (const int*)  d_in[1];
  const float* w_ih_f  = (const float*)d_in[2];
  const float* w_hh_f  = (const float*)d_in[3];
  const float* b_f     = (const float*)d_in[4];
  const float* w_ih_b  = (const float*)d_in[5];
  const float* w_hh_b  = (const float*)d_in[6];
  const float* b_b     = (const float*)d_in[7];
  const float* aw1     = (const float*)d_in[8];
  const float* ab1     = (const float*)d_in[9];
  const float* aw2     = (const float*)d_in[10];
  const float* ab2     = (const float*)d_in[11];
  const float* fc_w    = (const float*)d_in[12];
  const float* fc_b    = (const float*)d_in[13];
  const float* start_t = (const float*)d_in[14];
  const float* trans   = (const float*)d_in[15];
  const float* end_t   = (const float*)d_in[16];
  float* out = (float*)d_out;

  char* ws = (char*)d_ws;
  size_t off = 0;
  auto salloc = [&](size_t bytes) -> char* {
    char* p = ws + off;
    off = (off + bytes + 255) & ~(size_t)255;
    return p;
  };
  bf16_t* wc    = (bf16_t*)salloc((size_t)2 * G4 * KH * 2);        // 1.5 MB
  bf16_t* xbf   = (bf16_t*)salloc((size_t)B_ * T_ * D_ * 2);       // 16.8 MB
  bf16_t* lstm  = (bf16_t*)salloc((size_t)B_ * T_ * E2H * 2);      // 67 MB
  bf16_t* aw1b  = (bf16_t*)salloc((size_t)H_ * E2H * 2);           // 0.26 MB
  bf16_t* a_out = (bf16_t*)salloc((size_t)B_ * T_ * H_ * 2);       // 33.5 MB
  float*  scores= (float*) salloc((size_t)B_ * T_ * 4);            // 0.26 MB
  bf16_t* feat  = (bf16_t*)salloc((size_t)B_ * E2H * 2);           // 0.5 MB
  bf16_t* fcwb  = (bf16_t*)salloc((size_t)NFC * E2H * 2);          // 0.35 MB
  float*  em    = (float*) salloc((size_t)B_ * NFC * 4);           // 0.7 MB

  k_init<<<1, 64, 0, stream>>>(out);
  k_pack_wc<<<(2 * G4 * KH + 255) / 256, 256, 0, stream>>>(w_ih_f, w_hh_f, w_ih_b, w_hh_b, wc);
  k_cvt<<<(B_ * T_ * D_ + 255) / 256, 256, 0, stream>>>(x, xbf, B_ * T_ * D_);
  k_cvt<<<(H_ * E2H + 255) / 256, 256, 0, stream>>>(aw1, aw1b, H_ * E2H);
  k_cvt<<<(NFC * E2H + 255) / 256, 256, 0, stream>>>(fc_w, fcwb, NFC * E2H);

  k_lstm<<<dim3(B_ / 16, 2), 256, 0, stream>>>(xbf, wc, b_f, b_b, lstm);

  k_att_a<<<(B_ * T_ / 16) * 16 / 8, 256, 0, stream>>>(lstm, aw1b, ab1, a_out); // 8192 blocks
  k_score<<<B_ * T_ / 8, 256, 0, stream>>>(a_out, aw2, ab2, scores);            // 8192 blocks
  k_pool<<<B_, 128, 0, stream>>>(scores, lstm, feat);
  k_fc<<<(32 * NTFC) / 8, 256, 0, stream>>>(feat, fcwb, fc_b, em);              // 88 blocks
  k_crf<<<B_, 64, 0, stream>>>(em, labels, start_t, trans, end_t, out);
}
